// DeformableTokenEmbedding_68186900791514
// MI455X (gfx1250) — matrix operand order repaired and checked by reading, not compiled, not run on hardware
//
#include <hip/hip_runtime.h>

// MI455X / gfx1250 deformable token embedding.
// Three passes, heavy math via v_wmma_f32_16x16x32_bf16 (f32 accumulate).
// Reduction index ordered j = k*16 + c so the offset-GEMM A matrix is
// contiguous rows of x. bf16 packing via hardware v_cvt_pk_bf16_f32
// (inline asm: builtin not declared by this clang for gfx1250).

typedef __attribute__((ext_vector_type(16))) __bf16        v16bf;
typedef __attribute__((ext_vector_type(8)))  float         v8f;
typedef __attribute__((ext_vector_type(4)))  unsigned int  u32x4;
typedef __attribute__((ext_vector_type(4)))  float         f32x4;
typedef __attribute__((ext_vector_type(2)))  float         f32x2;

#define LN    36864
#define KN    24
#define CN    16
#define DN    512
#define LOUT  1536            /* LN / KN  */
#define RTOT  49152           /* 32*LOUT  */
#define CK    384             /* CN*KN    */

__device__ __forceinline__ unsigned short f2bf(float f) {
  union { float f; unsigned u; } v; v.f = f;
  unsigned u = v.u;
  u += 0x7FFFu + ((u >> 16) & 1u);   // round-to-nearest-even
  return (unsigned short)(u >> 16);
}

// Pack two f32 -> one dword of two bf16 (lo = a, hi = b), RNE, 1 VALU op.
__device__ __forceinline__ unsigned pack_bf16(float a, float b) {
  unsigned r;
  asm("v_cvt_pk_bf16_f32 %0, %1, %2" : "=v"(r) : "v"(a), "v"(b));
  return r;
}

union AFrag { v16bf v; u32x4 q[2]; unsigned u[8]; unsigned short s[16]; };

// ---------------------------------------------------------------------------
// Pass 0: repack weights w[col][c][k] (reduction j = k*16 + c) into per-lane
// WMMA B-fragment order, bf16.
//   col = lane&15 ; K = (e&7) + (e>=8 ? 16 : 0) + (lane>=16 ? 8 : 0)  for element e = 0..15
// ---------------------------------------------------------------------------
__global__ void prep_frag_kernel(const float* __restrict__ w,
                                 unsigned short* __restrict__ frag, int NT) {
  int t = blockIdx.x * 256 + threadIdx.x;
  int total = 12 * NT * 32 * 16;
  if (t >= total) return;
  int e    = t & 15;
  int lane = (t >> 4) & 31;
  int g    = t >> 9;            // kk*NT + nt
  int nt   = g % NT;
  int kk   = g / NT;
  int col  = nt * 16 + (lane & 15);
  int kidx = kk * 32 + (e & 7) + ((e >> 3) << 4) + ((lane >= 16) ? 8 : 0);
  int c    = kidx & 15;
  int k    = kidx >> 4;
  frag[t] = f2bf(w[(col * 16 + c) * 24 + k]);
}

// ---------------------------------------------------------------------------
// Pass 1: offset GEMM (49152 x 48, K=384) with WMMA bf16; then per (row,k)
// sampling index + folded bilinear weights -> workspace.
// 4 waves/block, 16 rows/wave, 64 rows/block, grid = 768.
// ---------------------------------------------------------------------------
__global__ __launch_bounds__(128)
void offsets_kernel(const float* __restrict__ x,
                    const unsigned short* __restrict__ woff_frag,
                    const float* __restrict__ b_off,
                    int* __restrict__ idx_ws,
                    float* __restrict__ wgt_ws) {
  const int lane = threadIdx.x & 31;
  const int wave = threadIdx.x >> 5;
  const int r0   = blockIdx.x * 64 + wave * 16;
  const int row  = r0 + (lane & 15);
  const int off8 = (lane & 16) ? 8 : 0;
  // 64-row blocks never straddle a batch (1536 % 64 == 0): block-constant l.
  const int lb0  = r0 % LOUT;

  v8f acc[3] = {};

#pragma unroll
  for (int kk = 0; kk < 12; ++kk) {
    // A fragment: two contiguous 8-float groups of row `row` of x.
    const float* pa = x + (size_t)row * CK + kk * 32 + off8;
    f32x4 g0 = *(const f32x4*)(pa +  0);
    f32x4 g1 = *(const f32x4*)(pa +  4);
    f32x4 g2 = *(const f32x4*)(pa + 16);
    f32x4 g3 = *(const f32x4*)(pa + 20);
    AFrag a;
    a.u[0] = pack_bf16(g0.x, g0.y);  a.u[1] = pack_bf16(g0.z, g0.w);
    a.u[2] = pack_bf16(g1.x, g1.y);  a.u[3] = pack_bf16(g1.z, g1.w);
    a.u[4] = pack_bf16(g2.x, g2.y);  a.u[5] = pack_bf16(g2.z, g2.w);
    a.u[6] = pack_bf16(g3.x, g3.y);  a.u[7] = pack_bf16(g3.z, g3.w);

#pragma unroll
    for (int nt = 0; nt < 3; ++nt) {
      const u32x4* bp =
          (const u32x4*)(woff_frag + (((size_t)kk * 3 + nt) * 32 + lane) * 16);
      AFrag bf; bf.q[0] = bp[0]; bf.q[1] = bp[1];
      acc[nt] = __builtin_amdgcn_wmma_f32_16x16x32_bf16(
          false, a.v, false, bf.v, (short)0, acc[nt], false, false);
    }
  }

  // C layout: VGPR v -> M = v + 8*(lane>=16), N = nt*16 + (lane&15).
  // Even N = dy, odd N = dx (torchvision channel layout), k = N/2.
  const int ln  = lane & 15;
  const int mhi = (lane & 16) ? 8 : 0;
#pragma unroll
  for (int nt = 0; nt < 3; ++nt) {
    const float bo = b_off[nt * 16 + ln];
#pragma unroll
    for (int v = 0; v < 8; ++v) {
      float val   = acc[nt][v] + bo;
      float other = __shfl_xor(val, 1);   // partner channel within (dy,dx) pair
      if ((ln & 1) == 0) {
        const int   k   = nt * 8 + (ln >> 1);
        const int   m   = v + mhi;
        const int   rr  = r0 + m;
        const int   l   = lb0 + m;
        const float dy  = val;
        const float dx  = other;
        const float px  = (float)(l * KN + k) + dx;
        const float wy  = fmaxf(0.f, 1.f - fabsf(dy));
        const bool  vld = (px > -1.f) && (px < (float)LN);
        const float x0  = floorf(px);
        const float lw  = px - x0;
        const int   i0  = (int)x0;
        const int   i1  = i0 + 1;
        const float m0  = (vld && i0 >= 0 && i0 < LN) ? 1.f : 0.f;
        const float m1  = (vld && i1 >= 0 && i1 < LN) ? 1.f : 0.f;
        const float w0  = (1.f - lw) * m0 * wy;
        const float w1  = lw * m1 * wy;
        const int   i0c = min(max(i0, 0), LN - 1);
        const int   o   = rr * KN + k;
        idx_ws[o] = i0c;
        *(f32x2*)(wgt_ws + 2 * o) = (f32x2){w0, w1};
      }
    }
  }
}

// ---------------------------------------------------------------------------
// Pass 2: gather val tile (32 rows x 384, bf16) into LDS (pitch 392 kills
// bank conflicts), then WMMA against frag-ordered w_def. 8 waves = 2 row
// groups x 4 col groups; block tile 32 rows x 256 cols. grid = (1536, 2).
// ---------------------------------------------------------------------------
__global__ __launch_bounds__(256)
void deform_gemm_kernel(const float* __restrict__ x,
                        const unsigned short* __restrict__ wdef_frag,
                        const float* __restrict__ b_def,
                        const int* __restrict__ idx_ws,
                        const float* __restrict__ wgt_ws,
                        float* __restrict__ out) {
  __shared__ unsigned short vsm[32 * 392];

  const int lane = threadIdx.x & 31;
  const int wave = threadIdx.x >> 5;
  const int rw   = wave >> 2;           // 0..1  row group
  const int cw   = wave & 3;            // 0..3  col group
  const int r0b  = blockIdx.x * 32;
  const int r0   = r0b + rw * 16;
  const int c0   = blockIdx.y * 256 + cw * 64;
  // 32-row blocks never straddle a batch (1536 % 32 == 0): block-constant b.
  const float* xb = x + (size_t)(r0b / LOUT) * LN * CN;

  // Build gathered val tile: 32*24 = 768 (row,k) pairs, 256 threads, 3 each.
  for (int p = threadIdx.x; p < 32 * KN; p += 256) {
    const int rl = p / KN;
    const int k  = p % KN;
    const int o  = (r0b + rl) * KN + k;
    const int i0 = idx_ws[o];
    const f32x2 w = *(const f32x2*)(wgt_ws + 2 * o);
    const f32x4* p0 = (const f32x4*)(xb + (size_t)i0 * CN);
    const int i1 = min(i0 + 1, LN - 1);
    const f32x4* p1 = (const f32x4*)(xb + (size_t)i1 * CN);
    unsigned* dst = (unsigned*)&vsm[rl * 392 + k * 16];
#pragma unroll
    for (int q = 0; q < 4; ++q) {
      f32x4 a0 = p0[q], a1 = p1[q];
      float c0v = a0.x * w.x + a1.x * w.y;
      float c1v = a0.y * w.x + a1.y * w.y;
      float c2v = a0.z * w.x + a1.z * w.y;
      float c3v = a0.w * w.x + a1.w * w.y;
      dst[q * 2 + 0] = pack_bf16(c0v, c1v);
      dst[q * 2 + 1] = pack_bf16(c2v, c3v);
    }
  }
  __syncthreads();

  const int off8 = (lane & 16) ? 8 : 0;
  const int mm   = lane & 15;
  v8f acc[4] = {};

#pragma unroll
  for (int kk = 0; kk < 12; ++kk) {
    const u32x4* ap =
        (const u32x4*)&vsm[(rw * 16 + mm) * 392 + kk * 32 + off8];
    AFrag a;
    a.q[0] = ap[0];     // K group 0 (8 bf16)
    a.q[1] = ap[2];     // K group 1 at +16 elements (+32 B)
#pragma unroll
    for (int nt = 0; nt < 4; ++nt) {
      const int ntg = (c0 >> 4) + nt;
      const u32x4* bp =
          (const u32x4*)(wdef_frag + (((size_t)kk * 32 + ntg) * 32 + lane) * 16);
      AFrag bf; bf.q[0] = bp[0]; bf.q[1] = bp[1];
      acc[nt] = __builtin_amdgcn_wmma_f32_16x16x32_bf16(
          false, a.v, false, bf.v, (short)0, acc[nt], false, false);
    }
  }

  // Epilogue: bias + store. C layout: M = v + 8*(lane>=16), N = lane&15.
  const int ln  = lane & 15;
  const int mhi = (lane & 16) ? 8 : 0;
#pragma unroll
  for (int nt = 0; nt < 4; ++nt) {
    const int col = c0 + nt * 16 + ln;
    const float bd = b_def[col];
#pragma unroll
    for (int v = 0; v < 8; ++v) {
      const int rr = r0 + v + mhi;
      out[(size_t)rr * DN + col] = acc[nt][v] + bd;
    }
  }
}

// ---------------------------------------------------------------------------
extern "C" void kernel_launch(void* const* d_in, const int* in_sizes, int n_in,
                              void* d_out, int out_size, void* d_ws, size_t ws_size,
                              hipStream_t stream) {
  const float* x     = (const float*)d_in[0];
  const float* w_off = (const float*)d_in[1];
  const float* b_off = (const float*)d_in[2];
  const float* w_def = (const float*)d_in[3];
  const float* b_def = (const float*)d_in[4];
  float* out = (float*)d_out;

  char* ws = (char*)d_ws;
  int*            idx_ws    = (int*)ws;                                 // 4,718,592 B
  float*          wgt_ws    = (float*)(ws + 4718592);                   // 9,437,184 B
  unsigned short* woff_frag = (unsigned short*)(ws + 14155776);         //    36,864 B
  unsigned short* wdef_frag = (unsigned short*)(ws + 14155776 + 36864); //   393,216 B

  prep_frag_kernel<<<(12 * 3 * 32 * 16 + 255) / 256, 256, 0, stream>>>(
      w_off, woff_frag, 3);
  prep_frag_kernel<<<(12 * 32 * 32 * 16 + 255) / 256, 256, 0, stream>>>(
      w_def, wdef_frag, 32);
  offsets_kernel<<<RTOT / 64, 128, 0, stream>>>(
      x, woff_frag, b_off, idx_ws, wgt_ws);
  deform_gemm_kernel<<<dim3(RTOT / 32, 2), 256, 0, stream>>>(
      x, wdef_frag, b_def, idx_ws, wgt_ws, out);
}
